// DHNSampler_70549132804599
// MI455X (gfx1250) — matrix operand order repaired and checked by reading, not compiled, not run on hardware
//
#include <hip/hip_runtime.h>
#include <hip/hip_bf16.h>

// Problem constants (from reference): B=4096 queries, Q=65536 queue rows, D=128, K=256
#define B_ 4096
#define Q_ 65536
#define D_ 128
#define K_ 256

typedef __attribute__((ext_vector_type(16))) __bf16 v16bf;
typedef __attribute__((ext_vector_type(8)))  __bf16 v8bf;
typedef __attribute__((ext_vector_type(8)))  float  v8f;

// Monotone order-preserving map: float -> uint (larger float => larger uint)
__device__ __forceinline__ unsigned monot(float f) {
    unsigned u = __float_as_uint(f);
    return (u & 0x80000000u) ? ~u : (u | 0x80000000u);
}

// ---------------------------------------------------------------------------
// One-shot fp32 -> bf16 conversion of the queue (done once, not per GEMM pass).
// 8 elements/thread; output is 16B-aligned v8bf stores.
// ---------------------------------------------------------------------------
__global__ void __launch_bounds__(256)
dhn_cvt(const float* __restrict__ src, __bf16* __restrict__ dst) {
    const size_t i = ((size_t)blockIdx.x * 256 + threadIdx.x) * 8;
    float4 f0 = *(const float4*)(src + i);
    float4 f1 = *(const float4*)(src + i + 4);
    v8bf o;
    o[0] = (__bf16)f0.x; o[1] = (__bf16)f0.y; o[2] = (__bf16)f0.z; o[3] = (__bf16)f0.w;
    o[4] = (__bf16)f1.x; o[5] = (__bf16)f1.y; o[6] = (__bf16)f1.z; o[7] = (__bf16)f1.w;
    *(v8bf*)(dst + i) = o;
}

// Build the four A fragments (kc = 0,32,64,96) once per wave from LDS.
// A 16x32 bf16 layout: lanes 0-15 -> K 0..15 contiguous; lanes 16-31 -> K 16..31.
// Both 8-element runs are 16B-aligned (row stride 136 bf16 = 272 B, 16 | 272).
__device__ __forceinline__ void load_afrags(const __bf16 (*Abf)[136], int lane,
                                            v16bf* afr) {
    const int half = lane >> 4;
    const int row  = lane & 15;
#pragma unroll
    for (int kc = 0; kc < 4; ++kc) {
        v8bf lo = *(const v8bf*)&Abf[row][kc * 32 + 16 * half];
        v8bf hi = *(const v8bf*)&Abf[row][kc * 32 + 16 * half + 8];
#pragma unroll
        for (int j = 0; j < 8; ++j) { afr[kc][j] = lo[j]; afr[kc][8 + j] = hi[j]; }
    }
}

// Compute a PAIR of adjacent 16x16 sim tiles (cols [n0,n0+16) and [n0+16,n0+32)),
// reusing the register-resident A fragments. B fragment per lane is K =
// kc*32 + 16*half + 0..15, contiguous bf16 in a queue row -> two b128 loads.
// 8 WMMAs in 4 independent accumulator chains.
__device__ __forceinline__ void sim_pair(const v16bf* afr,
                                         const __bf16* __restrict__ qbf,
                                         int n0, int lane, v8f& o0, v8f& o1) {
    const int half = lane >> 4;
    const int col  = lane & 15;
    const __bf16* b0p = qbf + (size_t)(n0 + col) * D_ + 16 * half;
    const __bf16* b1p = b0p + (size_t)16 * D_;
    v8f a0 = {}, a1 = {}, a2 = {}, a3 = {};
#pragma unroll
    for (int kc = 0; kc < 4; ++kc) {
        v16bf b0 = *(const v16bf*)(b0p + kc * 32);
        v16bf b1 = *(const v16bf*)(b1p + kc * 32);
        if (kc & 1) {
            a1 = __builtin_amdgcn_wmma_f32_16x16x32_bf16(false, afr[kc], false, b0,
                                                         (short)0, a1, false, false);
            a3 = __builtin_amdgcn_wmma_f32_16x16x32_bf16(false, afr[kc], false, b1,
                                                         (short)0, a3, false, false);
        } else {
            a0 = __builtin_amdgcn_wmma_f32_16x16x32_bf16(false, afr[kc], false, b0,
                                                         (short)0, a0, false, false);
            a2 = __builtin_amdgcn_wmma_f32_16x16x32_bf16(false, afr[kc], false, b1,
                                                         (short)0, a2, false, false);
        }
    }
#pragma unroll
    for (int r = 0; r < 8; ++r) { o0[r] = a0[r] + a1[r]; o1[r] = a2[r] + a3[r]; }
}

// ---------------------------------------------------------------------------
// Phase 1: per-row 2048-bucket histogram of monotone(sim) -> K-th threshold.
// One block per 16-query row-block; 8 waves split 2048 column tile-pairs.
// ---------------------------------------------------------------------------
__global__ void __launch_bounds__(256)
dhn_phase1(const float* __restrict__ query, const __bf16* __restrict__ qbf,
           int* __restrict__ thresh, int* __restrict__ base,
           int* __restrict__ cntA, int* __restrict__ cntB) {
    __shared__ __bf16 Abf[16][136];      // padded: conflict-free, 16B-aligned rows
    __shared__ int hist[16][2048];       // 128 KB of the 320 KB WGP LDS
    const int rb = blockIdx.x;

    for (int i = threadIdx.x; i < 16 * D_; i += 256) {
        int r = i >> 7, c = i & 127;
        Abf[r][c] = (__bf16)query[(size_t)(rb * 16 + r) * D_ + c];
    }
    for (int i = threadIdx.x; i < 16 * 2048; i += 256) ((int*)hist)[i] = 0;
    __syncthreads();

    const int wave = threadIdx.x >> 5, lane = threadIdx.x & 31;
    const int half = lane >> 4, col = lane & 15;
    v16bf afr[4];
    load_afrags(Abf, lane, afr);          // A lives in registers from here on

    for (int p = wave; p < Q_ / 32; p += 8) {
        const int n0 = p << 5;
        if (p + 8 < Q_ / 32)
            __builtin_prefetch(qbf + (size_t)(n0 + 256 + col) * D_, 0, 1);
        v8f s0, s1;
        sim_pair(afr, qbf, n0, lane, s0, s1);
#pragma unroll
        for (int r = 0; r < 8; ++r) {
            atomicAdd(&hist[r + 8 * half][monot(s0[r]) >> 21], 1);   // ds_add
            atomicAdd(&hist[r + 8 * half][monot(s1[r]) >> 21], 1);
        }
    }
    __syncthreads();

    if (threadIdx.x < 16) {
        const int row = threadIdx.x;
        int cum = 0, tb = 0;
        for (int bkt = 2047; bkt >= 0; --bkt) {
            int c = hist[row][bkt];
            if (cum + c >= K_) { tb = bkt; break; }
            cum += c;
        }
        const int g = rb * 16 + row;
        thresh[g] = tb;   // K-th value falls inside this bucket
        base[g]   = cum;  // # strictly-above-threshold-bucket elements (< K)
        cntA[g]   = 0;    // slot counters for phase 2
        cntB[g]   = 0;
    }
}

// ---------------------------------------------------------------------------
// Phase 2: recompute sim (identical fp path => identical values), scatter
// qualifying (idx, val) into per-row slots [0, K) via global atomics.
// grid (256, 4): blockIdx.y slices the 2048 tile-pairs for more parallelism.
// ---------------------------------------------------------------------------
__global__ void __launch_bounds__(256)
dhn_phase2(const float* __restrict__ query, const __bf16* __restrict__ qbf,
           const int* __restrict__ thresh, const int* __restrict__ base,
           int* __restrict__ cntA, int* __restrict__ cntB,
           int* __restrict__ idxout, float* __restrict__ valout) {
    __shared__ __bf16 Abf[16][136];
    const int rb = blockIdx.x;
    for (int i = threadIdx.x; i < 16 * D_; i += 256) {
        int r = i >> 7, c = i & 127;
        Abf[r][c] = (__bf16)query[(size_t)(rb * 16 + r) * D_ + c];
    }
    __syncthreads();

    const int wave = threadIdx.x >> 5, lane = threadIdx.x & 31;
    const int half = lane >> 4, col = lane & 15;
    v16bf afr[4];
    load_afrags(Abf, lane, afr);

    int tbv[8], bsv[8];
#pragma unroll
    for (int r = 0; r < 8; ++r) {
        int g = rb * 16 + r + 8 * half;
        tbv[r] = thresh[g];
        bsv[r] = base[g];
    }

    const int p0 = blockIdx.y * 512, p1 = p0 + 512;
    for (int p = p0 + wave; p < p1; p += 8) {
        const int n0 = p << 5;
        if (p + 8 < p1)
            __builtin_prefetch(qbf + (size_t)(n0 + 256 + col) * D_, 0, 1);
        v8f s0, s1;
        sim_pair(afr, qbf, n0, lane, s0, s1);
#pragma unroll
        for (int u = 0; u < 2; ++u) {
            const int n = n0 + 16 * u + col;
#pragma unroll
            for (int r = 0; r < 8; ++r) {
                float v = (u == 0) ? s0[r] : s1[r];
                int bkt = (int)(monot(v) >> 21);
                int g = rb * 16 + r + 8 * half;
                if (bkt > tbv[r]) {                       // definitely in top-K
                    int pos = atomicAdd(&cntA[g], 1);
                    idxout[(size_t)g * K_ + pos] = n;
                    valout[(size_t)g * K_ + pos] = v;
                } else if (bkt == tbv[r]) {               // threshold-bucket ties
                    int pos = bsv[r] + atomicAdd(&cntB[g], 1);
                    if (pos < K_) {
                        idxout[(size_t)g * K_ + pos] = n;
                        valout[(size_t)g * K_ + pos] = v;
                    }
                }
            }
        }
    }
}

// ---------------------------------------------------------------------------
// Phase 2.5: per-row bitonic sort of the 256 selected (val, idx) pairs,
// descending by val (index-ascending tie-break) => deterministic, matches
// reference top_k ordering. One 256-thread block per query row.
// ---------------------------------------------------------------------------
__global__ void __launch_bounds__(256)
dhn_sort(int* __restrict__ idx, const float* __restrict__ val) {
    __shared__ float sv[K_];
    __shared__ int   si[K_];
    const int g = blockIdx.x, t = threadIdx.x;
    sv[t] = val[(size_t)g * K_ + t];
    si[t] = idx[(size_t)g * K_ + t];
    __syncthreads();
    for (int k = 2; k <= K_; k <<= 1) {
        for (int j = k >> 1; j > 0; j >>= 1) {
            int ixj = t ^ j;
            if (ixj > t) {
                float va = sv[t], vb = sv[ixj];
                int   ia = si[t], ib = si[ixj];
                bool aFirst = (va > vb) || (va == vb && ia < ib);
                bool desc = ((t & k) == 0);
                if (desc ? !aFirst : aFirst) {
                    sv[t] = vb; sv[ixj] = va;
                    si[t] = ib; si[ixj] = ia;
                }
            }
            __syncthreads();
        }
    }
    idx[(size_t)g * K_ + t] = si[t];
}

// ---------------------------------------------------------------------------
// Phase 3: gather hard negatives: out[b,k,:] = queue[idx[b,k],:]
// One wave per (b,k) pair; 32 lanes x float4 = 128 floats per row.
// ---------------------------------------------------------------------------
__global__ void __launch_bounds__(256)
dhn_gather(const float* __restrict__ queue, const int* __restrict__ idx,
           float* __restrict__ out) {
    const int wave = threadIdx.x >> 5, lane = threadIdx.x & 31;
    const size_t p = (size_t)blockIdx.x * 8 + wave;
    const int id = idx[p];
    const float4* src = (const float4*)(queue + (size_t)id * D_);
    float4* dst = (float4*)(out + p * D_);
    dst[lane] = src[lane];
}

extern "C" void kernel_launch(void* const* d_in, const int* in_sizes, int n_in,
                              void* d_out, int out_size, void* d_ws, size_t ws_size,
                              hipStream_t stream) {
    const float* query = (const float*)d_in[0];   // [B, D] fp32
    const float* queue = (const float*)d_in[1];   // [Q, D] fp32, L2-normalized rows
    float* out = (float*)d_out;                   // [B, K, D] fp32

    // Workspace layout (~25 MB total)
    char* w = (char*)d_ws;
    __bf16* qbf = (__bf16*)w; w += (size_t)Q_ * D_ * sizeof(__bf16);  // 16 MB
    int* thresh = (int*)w;    w += (size_t)B_ * sizeof(int);
    int* base   = (int*)w;    w += (size_t)B_ * sizeof(int);
    int* cntA   = (int*)w;    w += (size_t)B_ * sizeof(int);
    int* cntB   = (int*)w;    w += (size_t)B_ * sizeof(int);
    int* idx    = (int*)w;    w += (size_t)B_ * K_ * sizeof(int);
    float* val  = (float*)w;

    dhn_cvt<<<(Q_ * D_) / (256 * 8), 256, 0, stream>>>(queue, qbf);
    dhn_phase1<<<B_ / 16, 256, 0, stream>>>(query, qbf, thresh, base, cntA, cntB);
    dhn_phase2<<<dim3(B_ / 16, 4), 256, 0, stream>>>(query, qbf, thresh, base,
                                                     cntA, cntB, idx, val);
    dhn_sort<<<B_, 256, 0, stream>>>(idx, val);
    dhn_gather<<<(B_ * K_) / 8, 256, 0, stream>>>(queue, idx, out);
}
